// Attention_71373766525161
// MI455X (gfx1250) — compile-verified, matrix operand order checked by reading, not run on hardware
//
#include <hip/hip_runtime.h>
#include <hip/hip_bf16.h>

// ---------------- CDNA5 WMMA types ----------------
typedef __attribute__((ext_vector_type(16))) __bf16 v16bf;
typedef __attribute__((ext_vector_type(8)))  float  v8f;
typedef __attribute__((ext_vector_type(4)))  unsigned int v4u;
typedef __attribute__((ext_vector_type(8)))  int v8i_t;
typedef __attribute__((ext_vector_type(4)))  int v4i_t;

#define B_SZ  4
#define L_SEQ 2048
#define EMB   768
#define NH    12
#define HD    64
#define MROWS (B_SZ * L_SEQ)   // 8192

__device__ __forceinline__ __bf16 f32_to_bf16(float f) {
  unsigned u = __builtin_bit_cast(unsigned, f);
  u += 0x7FFFu + ((u >> 16) & 1u);               // round-to-nearest-even
  unsigned short hs = (unsigned short)(u >> 16);
  return __builtin_bit_cast(__bf16, hs);
}

// 16-bit A/B fragment element -> K index mapping (ISA 7.12.2, 16-bit A 16x32)
__device__ __forceinline__ int kmap(int e, int half) {
  return (e & 7) + 8 * half + 16 * (e >> 3);
}

union AFrag {
  v16bf v;
  unsigned u[8];
  __bf16 h[16];
};

__device__ __forceinline__ v8f wmma_bf16(v16bf a, v16bf b, v8f c) {
  // (neg_a, A, neg_b, B, c_mod, C, reuse_a, reuse_b)
  return __builtin_amdgcn_wmma_f32_16x16x32_bf16(false, a, false, b, (short)0, c,
                                                 false, false);
}

// 16-lane xor-butterfly max using v_permlane16 (VALU, no LDS traffic).
// Selector nibbles encode lane i -> lane i^mask within each row of 16.
__device__ __forceinline__ float permlane_max16(float x) {
  int xi, yi;
  xi = __builtin_bit_cast(int, x);
  yi = __builtin_amdgcn_permlane16(xi, xi, (int)0x67452301, (int)0xEFCDAB89, false, false);
  x = fmaxf(x, __builtin_bit_cast(float, yi));
  xi = __builtin_bit_cast(int, x);
  yi = __builtin_amdgcn_permlane16(xi, xi, (int)0x54761032, (int)0xDCFE98BA, false, false);
  x = fmaxf(x, __builtin_bit_cast(float, yi));
  xi = __builtin_bit_cast(int, x);
  yi = __builtin_amdgcn_permlane16(xi, xi, (int)0x32107654, (int)0xBA98FEDC, false, false);
  x = fmaxf(x, __builtin_bit_cast(float, yi));
  xi = __builtin_bit_cast(int, x);
  yi = __builtin_amdgcn_permlane16(xi, xi, (int)0xFEDCBA98, (int)0x76543210, false, false);
  x = fmaxf(x, __builtin_bit_cast(float, yi));
  return x;
}

// ---------------- Stage 0: fp32 -> bf16 convert ----------------
__global__ void cvt_f32_bf16(const float* __restrict__ in, __bf16* __restrict__ out,
                             int n) {
  int i = blockIdx.x * blockDim.x + threadIdx.x;
  if (i < n) out[i] = f32_to_bf16(in[i]);
}

// ---------------- Stage 1/3: 768-K GEMM with bias ----------------
// OUTMODE 0: bf16 out scattered to [B, H, L, hd]   (QKV projection)
// OUTMODE 1: f32 out row-major [M, 768]            (output projection)
template <int OUTMODE>
__global__ __launch_bounds__(256) void gemm768(const __bf16* __restrict__ A,
                                               const __bf16* __restrict__ W,
                                               const float* __restrict__ bias,
                                               void* __restrict__ outp) {
  __shared__ __align__(16) __bf16 sWt[64 * 32];  // W tile transposed: [n][k]

  int wave = threadIdx.x >> 5;
  int lane = threadIdx.x & 31;
  int lrow = lane & 15, half = lane >> 4;
  int mbase = blockIdx.x * 128 + wave * 16;
  int nbase = blockIdx.y * 64;

  v8f acc[4] = {};

  for (int k0 = 0; k0 < EMB; k0 += 32) {
    // cooperative transposed stage of W[k0:k0+32, nbase:nbase+64] into LDS
    {
      int t   = threadIdx.x;
      int kk  = t >> 3;
      int off = (t & 7) * 8;
      uint4 raw = *(const uint4*)(W + (size_t)(k0 + kk) * EMB + nbase + off);
      const __bf16* wp = (const __bf16*)&raw;
#pragma unroll
      for (int j = 0; j < 8; ++j) sWt[(off + j) * 32 + kk] = wp[j];
    }
    __syncthreads();

    AFrag af;
#pragma unroll
    for (int e = 0; e < 16; e += 2)
      af.u[e >> 1] =
          *(const unsigned*)(A + (size_t)(mbase + lrow) * EMB + k0 + kmap(e, half));

#pragma unroll
    for (int seg = 0; seg < 4; ++seg) {
      AFrag bf;
#pragma unroll
      for (int e = 0; e < 16; e += 2)
        bf.u[e >> 1] = *(const unsigned*)(&sWt[(lrow + 16 * seg) * 32 + kmap(e, half)]);
      acc[seg] = wmma_bf16(af.v, bf.v, acc[seg]);
    }
    __syncthreads();
  }

#pragma unroll
  for (int seg = 0; seg < 4; ++seg) {
    int n = nbase + seg * 16 + lrow;
    float bval = bias[n];
#pragma unroll
    for (int v = 0; v < 8; ++v) {
      int m = mbase + v + 8 * half;
      float val = acc[seg][v] + bval;
      if (OUTMODE == 0) {
        int b = m >> 11, l = m & (L_SEQ - 1);
        int hidx = n >> 6, d = n & 63;
        ((__bf16*)outp)[(((size_t)b * NH + hidx) * L_SEQ + l) * HD + d] =
            f32_to_bf16(val);
      } else {
        ((float*)outp)[(size_t)m * EMB + n] = val;
      }
    }
  }
}

// ---------------- Stage 2: fused flash attention ----------------
// grid = (B*H, L/128); block = 256 (8 waves, 16 queries each)
__global__ __launch_bounds__(256) void flash_attn(const __bf16* __restrict__ Q,
                                                  const __bf16* __restrict__ K,
                                                  const __bf16* __restrict__ V,
                                                  __bf16* __restrict__ Oattn) {
  __shared__ __align__(16) __bf16 sK[32 * 64];      // [key][dim] (filled by TDM)
  __shared__ __align__(16) __bf16 sVt[64 * 32];     // [dim][key] (transposed)
  __shared__ __align__(16) __bf16 sP[8][16 * 32];   // per-wave P scratch

  int bh = blockIdx.x;  // 0..47
  int b = bh / NH, h = bh % NH;
  int qbase = blockIdx.y * 128;
  int wave = threadIdx.x >> 5;
  int lane = threadIdx.x & 31;
  int lrow = lane & 15, half = lane >> 4;

  const __bf16* Qh = Q + (size_t)bh * L_SEQ * HD;
  const __bf16* Kh = K + (size_t)bh * L_SEQ * HD;
  const __bf16* Vh = V + (size_t)bh * L_SEQ * HD;

  // Q tile (16 x 64) as two A-fragments, resident for the whole kernel
  AFrag qa[2];
  int qrow = qbase + wave * 16 + lrow;
#pragma unroll
  for (int kk = 0; kk < 2; ++kk)
#pragma unroll
    for (int e = 0; e < 16; e += 2)
      qa[kk].u[e >> 1] =
          *(const unsigned*)(Qh + (size_t)qrow * HD + kk * 32 + kmap(e, half));

  // all-ones bf16 B fragment: rowsum(P) via WMMA instead of lane reductions
  AFrag ones;
#pragma unroll
  for (int i = 0; i < 8; ++i) ones.u[i] = 0x3F803F80u;

  v8f oacc[4] = {};
  v8f osum = {};          // per-row running softmax denominator (all cols equal)
  float mrow[8];
#pragma unroll
  for (int v = 0; v < 8; ++v) mrow[v] = -3.0e38f;
  const float scale = 0.125f;  // 1/sqrt(64)

  // TDM descriptor pieces that are loop-invariant
  unsigned ldsK = (unsigned)(size_t)(void*)sK;              // LDS byte offset
  unsigned long long gK = (unsigned long long)(size_t)Kh;   // global VA

  for (int kt = 0; kt < L_SEQ / 32; ++kt) {
    // ---- K tile: async DMA via Tensor Data Mover (one issue per block) ----
    if (wave == 0) {
      unsigned long long ga = gK + (unsigned long long)kt * 32 * HD * sizeof(__bf16);
      v4u g0 = {1u,                                     // count=1 (valid D#)
                ldsK,                                   // lds_addr
                (unsigned)(ga & 0xFFFFFFFFu),           // global_addr[31:0]
                (unsigned)((ga >> 32) & 0x1FFFFFFu) |   // global_addr[56:32]
                    (2u << 30)};                        // type=2 ("image")
      v8i_t g1 = {(int)(1u << 16),          // wg_mask=0, data_size=1 (2 bytes)
                  (int)(64u << 16),         // tensor_dim0 = 64 (dims)
                  (int)(2048u << 16),       // tensor_dim1 = 2048 (keys)
                  (int)(64u << 16),         // tile_dim0 = 64
                  (int)32,                  // tile_dim1 = 32, tile_dim2 = 0
                  (int)64,                  // tensor_dim0_stride = 64
                  0, 0};
      v4i_t gz4 = {0, 0, 0, 0};
      v8i_t gz8 = {0, 0, 0, 0, 0, 0, 0, 0};
      __builtin_amdgcn_tensor_load_to_lds(g0, g1, gz4, gz4, gz8, 0);
    }

    // ---- V tile: cooperative transposed stage into LDS (overlaps with TDM) ----
    {
      int t = threadIdx.x;
      int key = t >> 3, off = (t & 7) * 8;
      uint4 vraw = *(const uint4*)(Vh + (size_t)(kt * 32 + key) * HD + off);
      const __bf16* vp = (const __bf16*)&vraw;
#pragma unroll
      for (int j = 0; j < 8; ++j) sVt[(off + j) * 32 + key] = vp[j];
    }
    if (kt + 1 < L_SEQ / 32)
      __builtin_prefetch(Vh + (size_t)((kt + 1) * 32) * HD, 0, 3);

    if (wave == 0) __builtin_amdgcn_s_wait_tensorcnt(0);
    __syncthreads();

    // S(16q x 32k) = Q (16x64) . K^T : 4 WMMAs
    v8f s0 = {}, s1 = {};
#pragma unroll
    for (int kk = 0; kk < 2; ++kk) {
      AFrag kb0, kb1;
#pragma unroll
      for (int e = 0; e < 16; e += 2) {
        int d = kk * 32 + kmap(e, half);
        kb0.u[e >> 1] = *(const unsigned*)(&sK[lrow * 64 + d]);
        kb1.u[e >> 1] = *(const unsigned*)(&sK[(lrow + 16) * 64 + d]);
      }
      s0 = wmma_bf16(qa[kk].v, kb0.v, s0);
      s1 = wmma_bf16(qa[kk].v, kb1.v, s1);
    }

    // online softmax: rows live in (v, half); keys across lanes 0..15 of group
#pragma unroll
    for (int v = 0; v < 8; ++v) {
      float a = s0[v] * scale;
      float c = s1[v] * scale;
      float mx = permlane_max16(fmaxf(a, c));      // row max, VALU butterfly
      float mnew = fmaxf(mrow[v], mx);
      float corr = __expf(mrow[v] - mnew);
      float p0 = __expf(a - mnew);
      float p1 = __expf(c - mnew);
      mrow[v] = mnew;
      oacc[0][v] *= corr;
      oacc[1][v] *= corr;
      oacc[2][v] *= corr;
      oacc[3][v] *= corr;
      osum[v] *= corr;
      // spill P (C layout) into per-wave LDS scratch as bf16
      int prow = v + 8 * half;
      sP[wave][prow * 32 + lrow]      = f32_to_bf16(p0);
      sP[wave][prow * 32 + lrow + 16] = f32_to_bf16(p1);
    }
    // LDS is in-order per wave; make the RAW explicit
    asm volatile("s_wait_dscnt 0" ::: "memory");

    // reload P in A-fragment layout
    AFrag pa;
#pragma unroll
    for (int e = 0; e < 16; e += 2)
      pa.u[e >> 1] = *(const unsigned*)(&sP[wave][lrow * 32 + kmap(e, half)]);

    // O(16 x 64) += P(16x32) . V(32x64) : 4 WMMAs
#pragma unroll
    for (int seg = 0; seg < 4; ++seg) {
      AFrag vb;
#pragma unroll
      for (int e = 0; e < 16; e += 2)
        vb.u[e >> 1] =
            *(const unsigned*)(&sVt[(lrow + 16 * seg) * 32 + kmap(e, half)]);
      oacc[seg] = wmma_bf16(pa.v, vb.v, oacc[seg]);
    }
    // row sums of P on the matrix pipe: osum += P . ones
    osum = wmma_bf16(pa.v, ones.v, osum);

    __syncthreads();
  }

  // epilogue: normalize and scatter to [B, L, H*hd] bf16 for the out-projection
#pragma unroll
  for (int seg = 0; seg < 4; ++seg) {
#pragma unroll
    for (int v = 0; v < 8; ++v) {
      int m = qbase + wave * 16 + v + 8 * half;
      int d = lrow + 16 * seg;
      float val = oacc[seg][v] / osum[v];
      Oattn[((size_t)b * L_SEQ + m) * EMB + h * HD + d] = f32_to_bf16(val);
    }
  }
}

// ---------------- host ----------------
extern "C" void kernel_launch(void* const* d_in, const int* in_sizes, int n_in,
                              void* d_out, int out_size, void* d_ws, size_t ws_size,
                              hipStream_t stream) {
  (void)in_sizes; (void)n_in; (void)out_size; (void)ws_size;
  const float* x  = (const float*)d_in[0];
  const float* Wq = (const float*)d_in[1];
  const float* bq = (const float*)d_in[2];
  const float* Wk = (const float*)d_in[3];
  const float* bk = (const float*)d_in[4];
  const float* Wv = (const float*)d_in[5];
  const float* bv = (const float*)d_in[6];
  const float* Wo = (const float*)d_in[7];
  const float* bo = (const float*)d_in[8];

  const size_t NX = (size_t)MROWS * EMB;  // 6,291,456
  const size_t NW = (size_t)EMB * EMB;    // 589,824

  __bf16* xb   = (__bf16*)d_ws;
  __bf16* wqb  = xb + NX;
  __bf16* wkb  = wqb + NW;
  __bf16* wvb  = wkb + NW;
  __bf16* wob  = wvb + NW;
  __bf16* Qb   = wob + NW;   // [B,H,L,hd]
  __bf16* Kb   = Qb + NX;
  __bf16* Vb   = Kb + NX;
  __bf16* Attn = Vb + NX;    // [B,L,H,hd] == [M, 768]

  cvt_f32_bf16<<<(int)((NX + 255) / 256), 256, 0, stream>>>(x, xb, (int)NX);
  cvt_f32_bf16<<<(int)((NW + 255) / 256), 256, 0, stream>>>(Wq, wqb, (int)NW);
  cvt_f32_bf16<<<(int)((NW + 255) / 256), 256, 0, stream>>>(Wk, wkb, (int)NW);
  cvt_f32_bf16<<<(int)((NW + 255) / 256), 256, 0, stream>>>(Wv, wvb, (int)NW);
  cvt_f32_bf16<<<(int)((NW + 255) / 256), 256, 0, stream>>>(Wo, wob, (int)NW);

  dim3 ggrid(MROWS / 128, EMB / 64);
  gemm768<0><<<ggrid, 256, 0, stream>>>(xb, wqb, bq, (void*)Qb);
  gemm768<0><<<ggrid, 256, 0, stream>>>(xb, wkb, bk, (void*)Kb);
  gemm768<0><<<ggrid, 256, 0, stream>>>(xb, wvb, bv, (void*)Vb);

  flash_attn<<<dim3(B_SZ * NH, L_SEQ / 128), 256, 0, stream>>>(Qb, Kb, Vb, Attn);

  gemm768<1><<<ggrid, 256, 0, stream>>>(Attn, wob, bo, d_out);
}
